// DecoderBlock_65335042506782
// MI455X (gfx1250) — matrix operand order repaired and checked by reading, not compiled, not run on hardware
//
#include <hip/hip_runtime.h>
#include <math.h>
#include <stdint.h>
#include <stddef.h>

// ---------------------------------------------------------------------------
// Types for CDNA5 WMMA (wave32, 16x16x32 f16 -> f32)
// ---------------------------------------------------------------------------
typedef __attribute__((ext_vector_type(16))) _Float16 v16h;
typedef __attribute__((ext_vector_type(8)))  _Float16 v8h;
typedef __attribute__((ext_vector_type(8)))  float    v8f;

#define NSEQ  512
#define DLOC  256
#define DPAIR 128
#define NHEAD 8
#define KDIM  32
#define HSZ   256   // 2*PS
#define PSZ   128

// ---------------------------------------------------------------------------
// Device helpers
// ---------------------------------------------------------------------------
__device__ __forceinline__ float gelu_f(float x) {
  const float c0 = 0.7978845608028654f;
  return 0.5f * x * (1.0f + tanhf(c0 * (x + 0.044715f * x * x * x)));
}

__device__ __forceinline__ float wave_sum(float x) {
#pragma unroll
  for (int off = 16; off > 0; off >>= 1) x += __shfl_xor(x, off, 32);
  return x;
}
__device__ __forceinline__ float wave_max(float x) {
#pragma unroll
  for (int off = 16; off > 0; off >>= 1) x = fmaxf(x, __shfl_xor(x, off, 32));
  return x;
}

// D = A x B + C.  A: 16x32 f16 row-major (lda elems), per ISA 16-bit A layout:
// lane m = l&15, khalf = l>>4; elems 0..7 -> k = khalf*8 + e; elems 8..15 -> +16.
__device__ __forceinline__ v16h load_afrag(const _Float16* base, int lda, int lane) {
  const int m = lane & 15, kh = lane >> 4;
  const _Float16* p = base + (size_t)m * lda + kh * 8;
  v8h x = *(const v8h*)p;
  v8h y = *(const v8h*)(p + 16);
  v16h a;
#pragma unroll
  for (int e = 0; e < 8; ++e) { a[e] = x[e]; a[e + 8] = y[e]; }
  return a;
}

// B stored pre-transposed Bt[n][k] (ldb = K): lane n = l&15, khalf = l>>4,
// elems 0..7 -> k = khalf*8 + e; elems 8..15 -> +16 (matches A).
__device__ __forceinline__ v16h load_bfrag(const _Float16* Bt, int ldb, int lane) {
  const int n = lane & 15, kh = lane >> 4;
  const _Float16* p = Bt + (size_t)n * ldb + kh * 8;
  v8h x = *(const v8h*)p;
  v8h y = *(const v8h*)(p + 16);
  v16h b;
#pragma unroll
  for (int e = 0; e < 8; ++e) { b[e] = x[e]; b[e + 8] = y[e]; }
  return b;
}

__device__ __forceinline__ v8f wmma_f16(v16h a, v16h b, v8f c) {
  return __builtin_amdgcn_wmma_f32_16x16x32_f16(false, a, false, b, (short)0, c,
                                                false, false);
}

// ---------------------------------------------------------------------------
// Weight transpose+convert: W[K][N] f32 -> Wt[N][K] f16
// ---------------------------------------------------------------------------
__global__ void k_transpose_f16(const float* __restrict__ W, _Float16* __restrict__ Wt,
                                int K, int N) {
  int idx = blockIdx.x * 256 + threadIdx.x;
  if (idx >= K * N) return;
  int n = idx / K, k = idx - n * K;
  Wt[idx] = (_Float16)W[(size_t)k * N + n];
}

// wb [128,8] -> padded Wt[16][128] (zero cols 8..15)
__global__ void k_transpose_pad_f16(const float* __restrict__ W, _Float16* __restrict__ Wt,
                                    int K, int Nsrc, int Npad) {
  int idx = blockIdx.x * 256 + threadIdx.x;
  if (idx >= K * Npad) return;
  int n = idx / K, k = idx - n * K;
  Wt[idx] = (_Float16)(n < Nsrc ? W[(size_t)k * Nsrc + n] : 0.0f);
}

// ---------------------------------------------------------------------------
// Row LayerNorm (one wave per row), optional affine, optional scale,
// optional f16 and/or f32 outputs. Safe for in-place f32 output.
// ---------------------------------------------------------------------------
__global__ void k_ln_rows(const float* __restrict__ X, const float* __restrict__ g,
                          const float* __restrict__ b, float scale,
                          _Float16* __restrict__ Yh, float* __restrict__ Yf,
                          int M, int D) {
  const int wave = threadIdx.x >> 5, lane = threadIdx.x & 31;
  const int row = blockIdx.x * (blockDim.x >> 5) + wave;
  if (row >= M) return;
  const float* x = X + (size_t)row * D;
  float s = 0.f, ss = 0.f;
  for (int c = lane; c < D; c += 32) { float v = x[c]; s += v; ss += v * v; }
  s = wave_sum(s); ss = wave_sum(ss);
  const float mu = s / (float)D;
  const float rs = rsqrtf(ss / (float)D - mu * mu + 1e-5f);
  for (int c = lane; c < D; c += 32) {
    float v = (x[c] - mu) * rs;
    if (g) v = v * g[c] + b[c];
    v *= scale;
    if (Yh) Yh[(size_t)row * D + c] = (_Float16)v;
    if (Yf) Yf[(size_t)row * D + c] = v;
  }
}

// ---------------------------------------------------------------------------
// Generic WMMA GEMM: one wave per (16-row, NT*16-col) tile.
// A f16 [M,K] row-major, Bt f16 [N][K]. Epilogues by template.
// ---------------------------------------------------------------------------
enum { EPI_F32 = 0, EPI_F16 = 1, EPI_BIAS_F32 = 2, EPI_RESID_F32 = 3 };

template <int NT, int EPI>
__global__ void k_gemm_wmma(const _Float16* __restrict__ A, const _Float16* __restrict__ Bt,
                            float* __restrict__ Of, _Float16* __restrict__ Oh,
                            const float* __restrict__ bias, const float* __restrict__ resid,
                            int K, int lda, int ldb, int ldo) {
  const int mt = blockIdx.x, ng = blockIdx.y, lane = threadIdx.x;
  const _Float16* Ar = A + (size_t)mt * 16 * lda;
  v8f acc[NT] = {};
  for (int k0 = 0; k0 < K; k0 += 32) {
    if (k0 + 32 < K) __builtin_prefetch(Ar + k0 + 32, 0, 3);
    v16h a = load_afrag(Ar + k0, lda, lane);
#pragma unroll
    for (int t = 0; t < NT; ++t) {
      const _Float16* bp = Bt + (size_t)((ng * NT + t) * 16) * ldb + k0;
      acc[t] = wmma_f16(a, load_bfrag(bp, ldb, lane), acc[t]);
    }
  }
  const int nn = lane & 15, mh = lane >> 4;
#pragma unroll
  for (int t = 0; t < NT; ++t) {
    const int col = (ng * NT + t) * 16 + nn;
#pragma unroll
    for (int r = 0; r < 8; ++r) {
      const size_t row = (size_t)mt * 16 + mh * 8 + r;
      float x = acc[t][r];
      if (EPI == EPI_BIAS_F32)  x += bias[col];
      if (EPI == EPI_RESID_F32) x += resid[row * ldo + col];
      if (EPI == EPI_F16) Oh[row * ldo + col] = (_Float16)x;
      else                Of[row * ldo + col] = x;
    }
  }
}

// H = gelu(A@Bg) * (A@Bd), f16 output.
template <int NT>
__global__ void k_gemm_dualgate(const _Float16* __restrict__ A,
                                const _Float16* __restrict__ Bg,
                                const _Float16* __restrict__ Bd,
                                _Float16* __restrict__ H,
                                int K, int lda, int ldb, int ldo) {
  const int mt = blockIdx.x, ng = blockIdx.y, lane = threadIdx.x;
  const _Float16* Ar = A + (size_t)mt * 16 * lda;
  v8f ag[NT] = {};
  v8f ad[NT] = {};
  for (int k0 = 0; k0 < K; k0 += 32) {
    v16h a = load_afrag(Ar + k0, lda, lane);
#pragma unroll
    for (int t = 0; t < NT; ++t) {
      const size_t nb = (size_t)((ng * NT + t) * 16) * ldb + k0;
      ag[t] = wmma_f16(a, load_bfrag(Bg + nb, ldb, lane), ag[t]);
      ad[t] = wmma_f16(a, load_bfrag(Bd + nb, ldb, lane), ad[t]);
    }
  }
  const int nn = lane & 15, mh = lane >> 4;
#pragma unroll
  for (int t = 0; t < NT; ++t) {
    const int col = (ng * NT + t) * 16 + nn;
#pragma unroll
    for (int r = 0; r < 8; ++r) {
      const size_t row = (size_t)mt * 16 + mh * 8 + r;
      H[row * ldo + col] = (_Float16)(gelu_f(ag[t][r]) * ad[t][r]);
    }
  }
}

// ---------------------------------------------------------------------------
// Attention: masked softmax over j for one (i,h) per block (512 threads).
// logits = q[i,h]·k[j,h] + bias[i,j,h]
// ---------------------------------------------------------------------------
__global__ void k_attn_softmax(const float* __restrict__ qn, const float* __restrict__ kn,
                               const float* __restrict__ biasb, const int* __restrict__ mask,
                               float* __restrict__ attn) {
  __shared__ float qs[KDIM];
  __shared__ float red[16];
  const int i = blockIdx.x >> 3, h = blockIdx.x & 7;
  const int j = threadIdx.x;
  if (j < KDIM) qs[j] = qn[(size_t)i * DLOC + h * KDIM + j];
  __syncthreads();
  const float* kj = kn + (size_t)j * DLOC + h * KDIM;
  float d = 0.f;
#pragma unroll
  for (int a = 0; a < KDIM; ++a) d += qs[a] * kj[a];
  d += biasb[((size_t)i * NSEQ + j) * 16 + h];
  const bool pm = (mask[i] != 0) && (mask[j] != 0);
  const float lg = pm ? d : -1e9f;

  float m = wave_max(lg);
  if ((j & 31) == 0) red[j >> 5] = m;
  __syncthreads();
  m = red[0];
#pragma unroll
  for (int w = 1; w < 16; ++w) m = fmaxf(m, red[w]);
  __syncthreads();

  const float e = __expf(lg - m);
  float s = wave_sum(e);
  if ((j & 31) == 0) red[j >> 5] = s;
  __syncthreads();
  s = 0.f;
#pragma unroll
  for (int w = 0; w < 16; ++w) s += red[w];

  attn[((size_t)i * NSEQ + j) * NHEAD + h] = pm ? e / s : 0.f;
}

// out[i,h,a] = sum_j attn[i,j,h] * v[j,h,a] ; out_pair[i,h,a] = sum_j attn[i,j,h]*pvals[i,j,a]
// concat16[i, 0:256] = out ; concat16[i, 256:512] = out_pair
__global__ void k_attn_out(const float* __restrict__ attn, const float* __restrict__ vmat,
                           const _Float16* __restrict__ pvals16,
                           _Float16* __restrict__ concat16) {
  const int i = blockIdx.x;
  const int t = threadIdx.x;          // 256 = (h,a)
  const int h = t >> 5, a = t & 31;
  float o = 0.f, op = 0.f;
  for (int j = 0; j < NSEQ; ++j) {
    const float w = attn[((size_t)i * NSEQ + j) * NHEAD + h];
    o  += w * vmat[(size_t)j * DLOC + h * KDIM + a];
    op += w * (float)pvals16[((size_t)i * NSEQ + j) * KDIM + a];
  }
  concat16[(size_t)i * 512 + h * KDIM + a]       = (_Float16)o;
  concat16[(size_t)i * 512 + 256 + h * KDIM + a] = (_Float16)op;
}

// ---------------------------------------------------------------------------
// LocalToPair reductions.
// left[i,c]  = sum_j pm * gelu(lg[i,c]+pg[i,j,c]) * (lv[j,c]+pv[i,j,c])
// right[j,c] = sum_i pm * gelu(rg[j,c]+pg[i,j,c]) * (rv[i,c]+pv[i,j,c])
// ---------------------------------------------------------------------------
__global__ void k_l2p_left(const float* __restrict__ lg, const float* __restrict__ lv,
                           const _Float16* __restrict__ pg16, const _Float16* __restrict__ pv16,
                           const int* __restrict__ mask, float* __restrict__ left) {
  const int i = blockIdx.x, c = threadIdx.x;   // 256 threads
  float acc = 0.f;
  if (mask[i]) {
    const float lgc = lg[(size_t)i * HSZ + c];
    for (int j = 0; j < NSEQ; ++j) {
      if (!mask[j]) continue;
      const size_t r = ((size_t)i * NSEQ + j) * HSZ + c;
      acc += gelu_f(lgc + (float)pg16[r]) * (lv[(size_t)j * HSZ + c] + (float)pv16[r]);
    }
  }
  left[(size_t)i * HSZ + c] = acc;
}

__global__ void k_l2p_right(const float* __restrict__ rg, const float* __restrict__ rv,
                            const _Float16* __restrict__ pg16, const _Float16* __restrict__ pv16,
                            const int* __restrict__ mask, float* __restrict__ right) {
  const int j = blockIdx.x, c = threadIdx.x;
  float acc = 0.f;
  if (mask[j]) {
    const float rgc = rg[(size_t)j * HSZ + c];
    for (int i = 0; i < NSEQ; ++i) {
      if (!mask[i]) continue;
      const size_t r = ((size_t)i * NSEQ + j) * HSZ + c;
      acc += gelu_f(rgc + (float)pg16[r]) * (rv[(size_t)i * HSZ + c] + (float)pv16[r]);
    }
  }
  right[(size_t)j * HSZ + c] = acc;
}

// ---------------------------------------------------------------------------
// Fused l2p output: per 16-row tile of flattened (i,j):
//   pseudo = LN(left[i]+right[j]) built in LDS (f16),
//   pair_out = pair_in + lnp_tile @ Wt[:,0:128] + pseudo @ Wt[:,128:384]
// Block = 128 threads (4 waves). WMMA A-frags come from LDS.
// ---------------------------------------------------------------------------
__global__ void k_l2p_out(const _Float16* __restrict__ lnp16,
                          const float* __restrict__ left, const float* __restrict__ right,
                          const _Float16* __restrict__ l2pwot,
                          const float* __restrict__ pair_in, float* __restrict__ pair_out) {
  __shared__ __align__(16) _Float16 ps[16 * HSZ];     // pseudo tile, 8KB
  __shared__ __align__(16) _Float16 lt[16 * DPAIR];   // ln_p tile, 4KB
  __shared__ float redS[128], redQ[128];
  __shared__ float mu_s[16], rs_s[16];

  const int r0 = blockIdx.x * 16;
  const int t = threadIdx.x;
  const int row = t >> 3, part = t & 7;        // 16 rows x 8 partials
  const int r = r0 + row, ii = r >> 9, jj = r & (NSEQ - 1);
  const float* L = left + (size_t)ii * HSZ;
  const float* R = right + (size_t)jj * HSZ;

  float s = 0.f, ss = 0.f;
#pragma unroll
  for (int e = 0; e < 32; ++e) {
    const float x = L[part * 32 + e] + R[part * 32 + e];
    s += x; ss += x * x;
  }
  redS[row * 8 + part] = s;
  redQ[row * 8 + part] = ss;
  __syncthreads();
  if (part == 0) {
    float S = 0.f, Q = 0.f;
#pragma unroll
    for (int p = 0; p < 8; ++p) { S += redS[row * 8 + p]; Q += redQ[row * 8 + p]; }
    const float mu = S / (float)HSZ;
    mu_s[row] = mu;
    rs_s[row] = rsqrtf(Q / (float)HSZ - mu * mu + 1e-5f);
  }
  __syncthreads();
  {
    const float mu = mu_s[row], rs = rs_s[row];
#pragma unroll
    for (int e = 0; e < 32; ++e) {
      const float x = L[part * 32 + e] + R[part * 32 + e];
      ps[row * HSZ + part * 32 + e] = (_Float16)((x - mu) * rs);
    }
  }
#pragma unroll
  for (int e = 0; e < 16; ++e) {               // 128 threads x 16 = 2048 halves
    const int idx = t * 16 + e;
    const int rr = idx >> 7, cc = idx & 127;
    lt[idx] = lnp16[(size_t)(r0 + rr) * DPAIR + cc];
  }
  __syncthreads();

  const int wv = t >> 5, lane = t & 31;
#pragma unroll
  for (int ct = 0; ct < 2; ++ct) {             // 4 waves x 2 tiles = 8 col tiles
    const int n0 = (wv * 2 + ct) * 16;
    v8f acc = {};
    for (int k0 = 0; k0 < DPAIR; k0 += 32)     // ln_p part (K=128)
      acc = wmma_f16(load_afrag(lt + k0, DPAIR, lane),
                     load_bfrag(l2pwot + (size_t)n0 * 384 + k0, 384, lane), acc);
    for (int k0 = 0; k0 < HSZ; k0 += 32)       // pseudo part (K=256)
      acc = wmma_f16(load_afrag(ps + k0, HSZ, lane),
                     load_bfrag(l2pwot + (size_t)n0 * 384 + 128 + k0, 384, lane), acc);
    const int nn = lane & 15, mh = lane >> 4;
#pragma unroll
    for (int rr = 0; rr < 8; ++rr) {
      const size_t orow = (size_t)(r0 + mh * 8 + rr);
      pair_out[orow * PSZ + n0 + nn] = pair_in[orow * PSZ + n0 + nn] + acc[rr];
    }
  }
}

// ---------------------------------------------------------------------------
// Host side
// ---------------------------------------------------------------------------
extern "C" void kernel_launch(void* const* d_in, const int* in_sizes, int n_in,
                              void* d_out, int out_size, void* d_ws, size_t ws_size,
                              hipStream_t stream) {
  (void)in_sizes; (void)n_in; (void)out_size; (void)ws_size;

  const float* local = (const float*)d_in[0];
  const float* pair  = (const float*)d_in[1];
  const int*   mask  = (const int*)d_in[2];
  const float* wq = (const float*)d_in[3];  const float* bq = (const float*)d_in[4];
  const float* wk = (const float*)d_in[5];  const float* bk = (const float*)d_in[6];
  const float* wv = (const float*)d_in[7];  const float* bv = (const float*)d_in[8];
  const float* wb = (const float*)d_in[9];
  const float* wp = (const float*)d_in[10];
  const float* wo = (const float*)d_in[11];
  const float* ln1_g = (const float*)d_in[12]; const float* ln1_b = (const float*)d_in[13];
  const float* u1_wg = (const float*)d_in[14]; const float* u1_wd = (const float*)d_in[15];
  const float* u1_wo = (const float*)d_in[16];
  const float* pg_w = (const float*)d_in[17]; const float* pv_w = (const float*)d_in[18];
  const float* lg_w = (const float*)d_in[19]; const float* lv_w = (const float*)d_in[20];
  const float* rg_w = (const float*)d_in[21]; const float* rv_w = (const float*)d_in[22];
  const float* l2p_wo = (const float*)d_in[23];
  const float* ln2_g = (const float*)d_in[24]; const float* ln2_b = (const float*)d_in[25];
  const float* u2_wg = (const float*)d_in[26]; const float* u2_wd = (const float*)d_in[27];
  const float* u2_wo = (const float*)d_in[28];

  float* outL = (float*)d_out;                    // [512,256]
  float* outP = outL + (size_t)NSEQ * DLOC;       // [512,512,128]

  const size_t NN = (size_t)NSEQ * NSEQ;
  char* W = (char*)d_ws;
  size_t cur = 0;
  auto take = [&](size_t bytes) -> void* {
    void* p = W + cur;
    cur = (cur + bytes + 255) & ~(size_t)255;
    return p;
  };

  // Big buffers
  _Float16* lnp16   = (_Float16*)take(NN * DPAIR * 2);   // 64 MB
  _Float16* pg16    = (_Float16*)take(NN * HSZ * 2);     // 128 MB (reused: ln2x16)
  _Float16* pv16    = (_Float16*)take(NN * HSZ * 2);     // 128 MB (reused: h2)
  float*    biasb   = (float*)take(NN * 16 * 4);         // 16 MB (padded H=16)
  _Float16* pvals16 = (_Float16*)take(NN * KDIM * 2);    // 16 MB
  float*    attnb   = (float*)take(NN * NHEAD * 4);      // 8 MB
  // Small buffers
  _Float16* lnl16   = (_Float16*)take((size_t)NSEQ * DLOC * 2);
  float*    qb      = (float*)take((size_t)NSEQ * DLOC * 4);
  float*    kb      = (float*)take((size_t)NSEQ * DLOC * 4);
  float*    vb      = (float*)take((size_t)NSEQ * DLOC * 4);
  _Float16* concat16= (_Float16*)take((size_t)NSEQ * 512 * 2);
  float*    local1  = (float*)take((size_t)NSEQ * DLOC * 4);
  _Float16* ln1x16  = (_Float16*)take((size_t)NSEQ * DLOC * 2);
  _Float16* h1_16   = (_Float16*)take((size_t)NSEQ * 512 * 2);
  _Float16* lnl2_16 = (_Float16*)take((size_t)NSEQ * DLOC * 2);
  float*    lgb     = (float*)take((size_t)NSEQ * HSZ * 4);
  float*    lvb     = (float*)take((size_t)NSEQ * HSZ * 4);
  float*    rgb     = (float*)take((size_t)NSEQ * HSZ * 4);
  float*    rvb     = (float*)take((size_t)NSEQ * HSZ * 4);
  float*    leftb   = (float*)take((size_t)NSEQ * HSZ * 4);
  float*    rightb  = (float*)take((size_t)NSEQ * HSZ * 4);
  // Transposed f16 weights (Bt[n][k])
  _Float16* wqt    = (_Float16*)take(256 * 256 * 2);
  _Float16* wkt    = (_Float16*)take(256 * 256 * 2);
  _Float16* wvt    = (_Float16*)take(256 * 256 * 2);
  _Float16* wbt    = (_Float16*)take(16 * 128 * 2);
  _Float16* wpt    = (_Float16*)take(32 * 128 * 2);
  _Float16* wot    = (_Float16*)take(256 * 512 * 2);
  _Float16* u1wgt  = (_Float16*)take(512 * 256 * 2);
  _Float16* u1wdt  = (_Float16*)take(512 * 256 * 2);
  _Float16* u1wot  = (_Float16*)take(256 * 512 * 2);
  _Float16* pgwt   = (_Float16*)take(256 * 128 * 2);
  _Float16* pvwt   = (_Float16*)take(256 * 128 * 2);
  _Float16* lgt    = (_Float16*)take(256 * 256 * 2);
  _Float16* lvt    = (_Float16*)take(256 * 256 * 2);
  _Float16* rgt    = (_Float16*)take(256 * 256 * 2);
  _Float16* rvt    = (_Float16*)take(256 * 256 * 2);
  _Float16* l2pwot = (_Float16*)take(128 * 384 * 2);
  _Float16* u2wgt  = (_Float16*)take(256 * 128 * 2);
  _Float16* u2wdt  = (_Float16*)take(256 * 128 * 2);
  _Float16* u2wot  = (_Float16*)take(128 * 256 * 2);
  // Aliased reuse (pg16/pv16 dead after l2p reductions + fused output)
  _Float16* ln2x16 = pg16;   // [NN,128] f16
  _Float16* h2_16  = pv16;   // [NN,256] f16

  auto TT = [&](const float* src, _Float16* dst, int K, int N) {
    int total = K * N;
    k_transpose_f16<<<(total + 255) / 256, 256, 0, stream>>>(src, dst, K, N);
  };

  // ---- 0. weight prep ----
  TT(wq, wqt, 256, 256); TT(wk, wkt, 256, 256); TT(wv, wvt, 256, 256);
  TT(wp, wpt, 128, 32);  TT(wo, wot, 512, 256);
  TT(u1_wg, u1wgt, 256, 512); TT(u1_wd, u1wdt, 256, 512); TT(u1_wo, u1wot, 512, 256);
  TT(pg_w, pgwt, 128, 256);   TT(pv_w, pvwt, 128, 256);
  TT(lg_w, lgt, 256, 256); TT(lv_w, lvt, 256, 256);
  TT(rg_w, rgt, 256, 256); TT(rv_w, rvt, 256, 256);
  TT(l2p_wo, l2pwot, 384, 128);
  TT(u2_wg, u2wgt, 128, 256); TT(u2_wd, u2wdt, 128, 256); TT(u2_wo, u2wot, 256, 128);
  k_transpose_pad_f16<<<(16 * 128 + 255) / 256, 256, 0, stream>>>(wb, wbt, 128, 8, 16);

  // ---- 1. LayerNorms of inputs ----
  k_ln_rows<<<NSEQ / 8, 256, 0, stream>>>(local, nullptr, nullptr, 1.f, lnl16, nullptr,
                                          NSEQ, DLOC);
  k_ln_rows<<<NN / 8, 256, 0, stream>>>(pair, nullptr, nullptr, 1.f, lnp16, nullptr,
                                        (int)NN, DPAIR);

  // ---- 2. q/k/v projections (WMMA) + per-head LN ----
  dim3 g_loc(NSEQ / 16, DLOC / 64);
  k_gemm_wmma<4, EPI_BIAS_F32><<<g_loc, 32, 0, stream>>>(lnl16, wqt, qb, nullptr, bq, nullptr,
                                                         256, 256, 256, 256);
  k_gemm_wmma<4, EPI_BIAS_F32><<<g_loc, 32, 0, stream>>>(lnl16, wkt, kb, nullptr, bk, nullptr,
                                                         256, 256, 256, 256);
  k_gemm_wmma<4, EPI_BIAS_F32><<<g_loc, 32, 0, stream>>>(lnl16, wvt, vb, nullptr, bv, nullptr,
                                                         256, 256, 256, 256);
  const float qscale = 1.0f / sqrtf((float)KDIM + 1e-6f);
  k_ln_rows<<<(NSEQ * NHEAD) / 8, 256, 0, stream>>>(qb, nullptr, nullptr, qscale,
                                                    nullptr, qb, NSEQ * NHEAD, KDIM);
  k_ln_rows<<<(NSEQ * NHEAD) / 8, 256, 0, stream>>>(kb, nullptr, nullptr, 1.f,
                                                    nullptr, kb, NSEQ * NHEAD, KDIM);

  // ---- 3. pair-derived bias (Nc=16 padded) and pvals (Nc=32), WMMA ----
  k_gemm_wmma<1, EPI_F32><<<dim3(NN / 16, 1), 32, 0, stream>>>(lnp16, wbt, biasb, nullptr,
                                                               nullptr, nullptr,
                                                               128, 128, 128, 16);
  k_gemm_wmma<2, EPI_F16><<<dim3(NN / 16, 1), 32, 0, stream>>>(lnp16, wpt, nullptr, pvals16,
                                                               nullptr, nullptr,
                                                               128, 128, 128, 32);

  // ---- 4. masked softmax + attention applies ----
  k_attn_softmax<<<NSEQ * NHEAD, NSEQ, 0, stream>>>(qb, kb, biasb, mask, attnb);
  k_attn_out<<<NSEQ, 256, 0, stream>>>(attnb, vb, pvals16, concat16);

  // ---- 5. local residual + Update(local) ----
  k_gemm_wmma<4, EPI_RESID_F32><<<g_loc, 32, 0, stream>>>(concat16, wot, local1, nullptr,
                                                          nullptr, local, 512, 512, 512, 256);
  k_ln_rows<<<NSEQ / 8, 256, 0, stream>>>(local1, ln1_g, ln1_b, 1.f, ln1x16, nullptr,
                                          NSEQ, DLOC);
  k_gemm_dualgate<4><<<dim3(NSEQ / 16, 512 / 64), 32, 0, stream>>>(ln1x16, u1wgt, u1wdt,
                                                                   h1_16, 256, 256, 256, 512);
  k_gemm_wmma<4, EPI_RESID_F32><<<g_loc, 32, 0, stream>>>(h1_16, u1wot, outL, nullptr,
                                                          nullptr, local1, 512, 512, 512, 256);

  // ---- 6. LocalToPair ----
  k_ln_rows<<<NSEQ / 8, 256, 0, stream>>>(outL, nullptr, nullptr, 1.f, lnl2_16, nullptr,
                                          NSEQ, DLOC);
  k_gemm_wmma<4, EPI_F32><<<g_loc, 32, 0, stream>>>(lnl2_16, lgt, lgb, nullptr, nullptr,
                                                    nullptr, 256, 256, 256, 256);
  k_gemm_wmma<4, EPI_F32><<<g_loc, 32, 0, stream>>>(lnl2_16, lvt, lvb, nullptr, nullptr,
                                                    nullptr, 256, 256, 256, 256);
  k_gemm_wmma<4, EPI_F32><<<g_loc, 32, 0, stream>>>(lnl2_16, rgt, rgb, nullptr, nullptr,
                                                    nullptr, 256, 256, 256, 256);
  k_gemm_wmma<4, EPI_F32><<<g_loc, 32, 0, stream>>>(lnl2_16, rvt, rvb, nullptr, nullptr,
                                                    nullptr, 256, 256, 256, 256);
  dim3 g_pair(NN / 16, HSZ / 64);
  k_gemm_wmma<4, EPI_F16><<<g_pair, 32, 0, stream>>>(lnp16, pgwt, nullptr, pg16, nullptr,
                                                     nullptr, 128, 128, 128, 256);
  k_gemm_wmma<4, EPI_F16><<<g_pair, 32, 0, stream>>>(lnp16, pvwt, nullptr, pv16, nullptr,
                                                     nullptr, 128, 128, 128, 256);
  k_l2p_left<<<NSEQ, HSZ, 0, stream>>>(lgb, lvb, pg16, pv16, mask, leftb);
  k_l2p_right<<<NSEQ, HSZ, 0, stream>>>(rgb, rvb, pg16, pv16, mask, rightb);
  // fused pseudo-LN + concat GEMM + pair residual -> outP
  k_l2p_out<<<NN / 16, 128, 0, stream>>>(lnp16, leftb, rightb, l2pwot, pair, outP);

  // ---- 7. Update(pair), in-place residual on outP ----
  k_ln_rows<<<NN / 8, 256, 0, stream>>>(outP, ln2_g, ln2_b, 1.f, ln2x16, nullptr,
                                        (int)NN, DPAIR);
  k_gemm_dualgate<4><<<dim3(NN / 16, HSZ / 64), 32, 0, stream>>>(ln2x16, u2wgt, u2wdt,
                                                                 h2_16, 128, 128, 128, 256);
  k_gemm_wmma<4, EPI_RESID_F32><<<dim3(NN / 16, PSZ / 64), 32, 0, stream>>>(
      h2_16, u2wot, outP, nullptr, nullptr, outP, 256, 256, 256, 128);
}
